// NTMHeadBase_19387482374144
// MI455X (gfx1250) — matrix operand order repaired and checked by reading, not compiled, not run on hardware
//
#include <hip/hip_runtime.h>
#include <math.h>

// CDNA5 / gfx1250: wave32, WMMA bf16 16x16x32 with f32 accumulate.
typedef __bf16 v16bf __attribute__((ext_vector_type(16)));
typedef float  v8f   __attribute__((ext_vector_type(8)));

#define EPSF 1e-16f
#define CDIM 128   // feature dim (fixed by reference)
#define TM   16    // batch rows per workgroup (one WMMA M-tile)
#define KSTR 132   // LDS k-tile row stride (pad: avoid bank conflicts)

__device__ __forceinline__ float softplus_f(float x) {
  return (x > 20.0f) ? x : log1pf(expf(x));
}

__device__ __forceinline__ float waveReduceSum(float v) {
#pragma unroll
  for (int o = 16; o >= 1; o >>= 1) v += __shfl_xor(v, o, 32);
  return v;
}

__device__ __forceinline__ float waveReduceMax(float v) {
#pragma unroll
  for (int o = 16; o >= 1; o >>= 1) v = fmaxf(v, __shfl_xor(v, o, 32));
  return v;
}

// Split a float into bf16 hi + bf16 lo (hi+lo ~ x to ~2^-18 rel).
struct Bf16Pair { __bf16 hi, lo; };
__device__ __forceinline__ Bf16Pair bf16_split(float x) {
  Bf16Pair p;
  p.hi = (__bf16)x;
  p.lo = (__bf16)(x - (float)p.hi);
  return p;
}

// ---- one-time prep: split M into packed bf16 hi/lo + row norms (in d_ws) ----
__global__ __launch_bounds__(128) void ntm_prep_kernel(
    const float* __restrict__ M, __bf16* __restrict__ mhi,
    __bf16* __restrict__ mlo, float* __restrict__ mnorm) {
  const int n = blockIdx.x;
  const int c = threadIdx.x;  // CDIM threads
  const size_t idx = (size_t)n * CDIM + c;
  float x = M[idx];
  Bf16Pair p = bf16_split(x);
  mhi[idx] = p.hi;
  mlo[idx] = p.lo;
  float ss = waveReduceSum(x * x);
  __shared__ float red[4];
  if ((c & 31) == 0) red[c >> 5] = ss;
  __syncthreads();
  if (c == 0) mnorm[n] = sqrtf(red[0] + red[1] + red[2] + red[3]);
}

template <bool PRE>
__global__ __launch_bounds__(256) void ntm_head_kernel(
    const float* __restrict__ kin, const float* __restrict__ beta_in,
    const float* __restrict__ g_in, const float* __restrict__ s_in,
    const float* __restrict__ gamma_in, const float* __restrict__ wprev,
    const float* __restrict__ M, const __bf16* __restrict__ mhi,
    const __bf16* __restrict__ mlo, const float* __restrict__ mnorm,
    float* __restrict__ out, int B, int N) {
  const int WSTR = N + 4;  // padded LDS row stride for the score tile
  extern __shared__ float smem[];
  float* wbuf  = smem;                  // [TM][WSTR]  scores -> wc -> wg -> ws
  float* kbuf  = wbuf + TM * WSTR;      // [TM][KSTR]  k tile (f32)
  float* knorm = kbuf + TM * KSTR;      // [TM]
  float* betaS = knorm + TM;            // [TM]
  float* gS    = betaS + TM;            // [TM]
  float* gamS  = gS + TM;               // [TM]
  float* s0S   = gamS + TM;             // [TM]
  float* s1S   = s0S + TM;              // [TM]
  float* s2S   = s1S + TM;              // [TM]

  const int tid  = threadIdx.x;
  const int lane = tid & 31;
  const int wave = tid >> 5;
  const int b0   = blockIdx.x * TM;
  const int r16  = lane & 15;  // N (or M) index within 16-wide tile
  const int h    = lane >> 4;  // half-wave select

  // ---- stage k tile into LDS (coalesced) ----
  for (int i = tid; i < TM * CDIM; i += 256) {
    int r = i >> 7, c = i & (CDIM - 1);
    kbuf[r * KSTR + c] = kin[(size_t)(b0 + r) * CDIM + c];
  }
  __syncthreads();

  // ---- prefetch w_prev rows for this block (hide HBM latency behind GEMM) ----
  {
    const float* wp0 = wprev + (size_t)b0 * N;
    for (int i = tid; i < (TM * N) / 32; i += 256)
      __builtin_prefetch(wp0 + (size_t)i * 32, 0, 1);
  }

  // ---- per-row head activations + ||k|| (threads 0..15) ----
  if (tid < TM) {
    const int b = b0 + tid;
    float ss = 0.0f;
    const float* kr = &kbuf[tid * KSTR];
#pragma unroll 8
    for (int c = 0; c < CDIM; ++c) { float v = kr[c]; ss += v * v; }
    knorm[tid] = sqrtf(ss);
    betaS[tid] = softplus_f(beta_in[b]);
    gS[tid]    = 1.0f / (1.0f + expf(-g_in[b]));
    gamS[tid]  = 1.0f + softplus_f(gamma_in[b]);
    float x0 = s_in[b * 3 + 0], x1 = s_in[b * 3 + 1], x2 = s_in[b * 3 + 2];
    float mx = fmaxf(x0, fmaxf(x1, x2));
    float e0 = expf(x0 - mx), e1 = expf(x1 - mx), e2 = expf(x2 - mx);
    float inv = 1.0f / (e0 + e1 + e2);
    s0S[tid] = e0 * inv; s1S[tid] = e1 * inv; s2S[tid] = e2 * inv;
  }

  // ---- build A fragments (k rows, split hi/lo) — K order used here (same as B):
  //      lane half h: elems 0..15 -> K = 16h+0..15
  v16bf ahi[4], alo[4];
  {
    const float* krow = &kbuf[r16 * KSTR];
#pragma unroll
    for (int kc = 0; kc < 4; ++kc) {
      const int base = 32 * kc + 16 * h;
#pragma unroll
      for (int j = 0; j < 8; ++j) {
        Bf16Pair p0 = bf16_split(krow[base + j]);
        Bf16Pair p1 = bf16_split(krow[base + 8 + j]);
        ahi[kc][j]     = p0.hi;  alo[kc][j]     = p0.lo;
        ahi[kc][j + 8] = p1.hi;  alo[kc][j + 8] = p1.lo;
      }
    }
  }
  __syncthreads();  // knorm / activations ready

  float kn_l[8], bt_l[8];
#pragma unroll
  for (int r8 = 0; r8 < 8; ++r8) {
    kn_l[r8] = knorm[r8 + 8 * h];
    bt_l[r8] = betaS[r8 + 8 * h];
  }

  // ---- GEMM over N tiles: sim = k . M^T with compensated bf16 WMMA ----
  const int nTiles = N >> 4;
  for (int t = wave; t < nTiles; t += 8) {
    const int n0 = t << 4;
    const int nrow = n0 + r16;  // B frag: row n per lane
    v8f acc = {};
    float mn;
    if constexpr (PRE) {
      // Pre-split bf16 B fragments: pure vector loads, zero in-loop convert.
      const __bf16* hrow = mhi + (size_t)nrow * CDIM;
      const __bf16* lrow = mlo + (size_t)nrow * CDIM;
      mn = mnorm[nrow];
#pragma unroll
      for (int kc = 0; kc < 4; ++kc) {
        // B frag, same K order as A: lane half h holds contiguous K = 16h .. 16h+15
        const int koff = 32 * kc + 16 * h;
        v16bf bhi = *reinterpret_cast<const v16bf*>(hrow + koff);
        v16bf blo = *reinterpret_cast<const v16bf*>(lrow + koff);
        acc = __builtin_amdgcn_wmma_f32_16x16x32_bf16(
            false, ahi[kc], false, bhi, (short)0, acc, false, false);
        acc = __builtin_amdgcn_wmma_f32_16x16x32_bf16(
            false, ahi[kc], false, blo, (short)0, acc, false, false);
        acc = __builtin_amdgcn_wmma_f32_16x16x32_bf16(
            false, alo[kc], false, bhi, (short)0, acc, false, false);
      }
    } else {
      // Fallback: split M in-loop from f32 (no workspace needed).
      const float* mrow = M + (size_t)nrow * CDIM;
      float msq = 0.0f;
#pragma unroll
      for (int kc = 0; kc < 4; ++kc) {
        const int koff = 32 * kc + 16 * h;
        float4 q[4];
#pragma unroll
        for (int qq = 0; qq < 4; ++qq)
          q[qq] = reinterpret_cast<const float4*>(mrow + koff)[qq];
        const float* sv = reinterpret_cast<const float*>(q);
        v16bf bhi, blo;
#pragma unroll
        for (int j = 0; j < 16; ++j) {
          float x = sv[j];
          msq += x * x;
          Bf16Pair p = bf16_split(x);
          bhi[j] = p.hi;
          blo[j] = p.lo;
        }
        acc = __builtin_amdgcn_wmma_f32_16x16x32_bf16(
            false, ahi[kc], false, bhi, (short)0, acc, false, false);
        acc = __builtin_amdgcn_wmma_f32_16x16x32_bf16(
            false, ahi[kc], false, blo, (short)0, acc, false, false);
        acc = __builtin_amdgcn_wmma_f32_16x16x32_bf16(
            false, alo[kc], false, bhi, (short)0, acc, false, false);
      }
      // lane and lane^16 together cover all 128 K values of row n
      msq += __shfl_xor(msq, 16, 32);
      mn = sqrtf(msq);
    }
#pragma unroll
    for (int r8 = 0; r8 < 8; ++r8) {
      // D layout: VGPR r8, half h -> M = r8 + 8h, N = lane&15
      float sim = acc[r8] / (kn_l[r8] * mn + EPSF);
      wbuf[(r8 + 8 * h) * WSTR + nrow] = bt_l[r8] * sim;
    }
  }
  __syncthreads();  // full score tile resident in LDS

  // ---- per-row pipeline (each row wholly owned by one wave) ----
  const int nm1 = N - 1;
  const int it  = N / 32;
  for (int rr = 0; rr < 2; ++rr) {
    const int row = wave + rr * 8;
    const int b   = b0 + row;
    float* wr = &wbuf[row * WSTR];

    // softmax over N: max
    float mx = -3.402823466e38f;
    for (int n = lane; n < N; n += 32) mx = fmaxf(mx, wr[n]);
    mx = waveReduceMax(mx);
    // exp + sum (store exp in place)
    float sum = 0.0f;
    for (int n = lane; n < N; n += 32) {
      float e = expf(wr[n] - mx);
      wr[n] = e;
      sum += e;
    }
    sum = waveReduceSum(sum);
    const float invsum = 1.0f / sum;

    // interpolate with w_prev (coalesced global read), store w_g in place
    const float gg = gS[row];
    const float* wp = wprev + (size_t)b * N;
    for (int n = lane; n < N; n += 32)
      wr[n] = gg * (wr[n] * invsum) + (1.0f - gg) * wp[n];

    // circular 3-tap shift + sharpen, in place with lane31->lane0 carry
    const float s0v = s0S[row], s1v = s1S[row], s2v = s2S[row];
    const float gam = gamS[row];
    const float first = wr[0];    // original wg[0] (overwritten at iter 0)
    float carry = wr[nm1];        // original wg[N-1] for lane0 @ iter 0
    float ssum = 0.0f;
    for (int j = 0; j < it; ++j) {
      const int n = j * 32 + lane;
      float v0  = wr[n];
      float vm1 = (lane == 0) ? carry : wr[n - 1];
      float vp1 = wr[(n + 1) & nm1];
      if (j == it - 1 && lane == 31) vp1 = first;  // wrap reads original wg[0]
      carry = __shfl(v0, 31, 32);
      float wt = s0v * vm1 + s1v * v0 + s2v * vp1;
      float ws = (wt > 0.0f) ? expf(gam * logf(wt)) : 0.0f;
      ssum += ws;
      wr[n] = ws;
    }
    ssum = waveReduceSum(ssum);
    const float invs = 1.0f / (ssum + EPSF);

    float* orow = out + (size_t)b * N;
    for (int n = lane; n < N; n += 32) orow[n] = wr[n] * invs;
  }
}

extern "C" void kernel_launch(void* const* d_in, const int* in_sizes, int n_in,
                              void* d_out, int out_size, void* d_ws,
                              size_t ws_size, hipStream_t stream) {
  (void)n_in; (void)out_size;
  const float* k     = (const float*)d_in[0];
  const float* beta  = (const float*)d_in[1];
  const float* g     = (const float*)d_in[2];
  const float* s     = (const float*)d_in[3];
  const float* gamma = (const float*)d_in[4];
  const float* wprev = (const float*)d_in[5];
  const float* M     = (const float*)d_in[6];
  float* out = (float*)d_out;

  const int C = CDIM;
  const int B = in_sizes[0] / C;  // k is [B,C]
  const int N = in_sizes[6] / C;  // M is [N,C]
  const int wstr = N + 4;
  const size_t shbytes =
      (size_t)(TM * wstr + TM * KSTR + 7 * TM) * sizeof(float);

  // Workspace layout: mhi [N*C] bf16 | mlo [N*C] bf16 | mnorm [N] f32
  const size_t need = (size_t)N * C * 2 * sizeof(__bf16) + (size_t)N * sizeof(float);
  const bool pre = (d_ws != nullptr) && (ws_size >= need);

  if (pre) {
    __bf16* mhi = (__bf16*)d_ws;
    __bf16* mlo = mhi + (size_t)N * C;
    float* mnorm = (float*)(mlo + (size_t)N * C);
    ntm_prep_kernel<<<N, CDIM, 0, stream>>>(M, mhi, mlo, mnorm);
    (void)hipFuncSetAttribute((const void*)ntm_head_kernel<true>,
                              hipFuncAttributeMaxDynamicSharedMemorySize,
                              (int)shbytes);
    ntm_head_kernel<true><<<B / TM, 256, shbytes, stream>>>(
        k, beta, g, s, gamma, wprev, M, mhi, mlo, mnorm, out, B, N);
  } else {
    (void)hipFuncSetAttribute((const void*)ntm_head_kernel<false>,
                              hipFuncAttributeMaxDynamicSharedMemorySize,
                              (int)shbytes);
    ntm_head_kernel<false><<<B / TM, 256, shbytes, stream>>>(
        k, beta, g, s, gamma, wprev, M, nullptr, nullptr, nullptr, out, B, N);
  }
}
